// TriangleAttentionStartingNode_62663572848696
// MI455X (gfx1250) — matrix operand order repaired and checked by reading, not compiled, not run on hardware
//
#include <hip/hip_runtime.h>
#include <hip/hip_bf16.h>
#include <math.h>

// ---------------------------------------------------------------------------
// TriangleAttentionStartingNode for gfx1250 (MI455X, wave32, WMMA).
// All GEMMs via v_wmma_f32_16x16x32_f16; all intermediates stored f16 in
// WMMA-fragment-friendly layouts so fragments load as contiguous b128 pairs.
// ---------------------------------------------------------------------------

#define NSEQ   256
#define CIN    128
#define CATT   32
#define NH     4
#define LN_EPS 1e-5f

typedef __attribute__((ext_vector_type(16))) _Float16 v16h;   // 32 bytes
typedef __attribute__((ext_vector_type(8)))  float    v8f;

__device__ __forceinline__ v8f wmma16(v16h a, v16h b, v8f c) {
    // (neg_a, A, neg_b, B, c_mod, C, reuse_a, reuse_b)
    return __builtin_amdgcn_wmma_f32_16x16x32_f16(false, a, false, b,
                                                  (short)0, c, false, false);
}

// 16-bit A-fragment order used here (k contiguous per lane half): element (m, k) of a
// 16 x 32K tile lives at F[(kk*32 + lane)*16 + t] with
//   kk = k>>5, kr = k&31, lane = m + (kr>>4)*16, t = kr&15
// so lane L's 16 halves are contiguous -> one aligned 32B read per fragment.
__device__ __forceinline__ int frag_idx(int m, int k) {
    const int kk = k >> 5, kr = k & 31;
    const int lane = m + ((kr >> 4) << 4);
    const int t = kr & 15;
    return (kk * 32 + lane) * 16 + t;
}

// B fragment from f16 weights stored transposed/contiguous in K:
// lane l holds column n = l&15; K = kb + (l<16 ? 0..15 : 16..31) -> contiguous.
__device__ __forceinline__ v16h load_b_frag_kcontig(const _Float16* __restrict__ Wt,
                                                    int ldk, int kb, int n, int lane) {
    const int kbase = kb + ((lane < 16) ? 0 : 16);
    return *(const v16h*)(Wt + (size_t)n * ldk + kbase);   // 32B aligned
}

// ---------------------------------------------------------------------------
// Stage 0: convert + transpose weights to f16, K-contiguous ([n][k]).
//   Wqkv (128x384) -> Wqt (384x128); Wgate (128x128) -> Wgt; Wout -> Wot.
// ---------------------------------------------------------------------------
__global__ __launch_bounds__(256) void prep_weights_kernel(
    const float* __restrict__ Wq, const float* __restrict__ Wg,
    const float* __restrict__ Wo,
    _Float16* __restrict__ Wqt, _Float16* __restrict__ Wgt,
    _Float16* __restrict__ Wot)
{
    const int t = blockIdx.x * 256 + threadIdx.x;
    if (t < 384 * CIN) {
        const int n = t >> 7, k = t & 127;
        Wqt[t] = (_Float16)Wq[k * 384 + n];
    } else if (t < (384 + 128) * CIN) {
        const int u = t - 384 * CIN, n = u >> 7, k = u & 127;
        Wgt[u] = (_Float16)Wg[k * 128 + n];
    } else if (t < (384 + 256) * CIN) {
        const int u = t - 512 * CIN, n = u >> 7, k = u & 127;
        Wot[u] = (_Float16)Wo[k * 128 + n];
    }
}

// ---------------------------------------------------------------------------
// Stage 1: fused LayerNorm + qkv / bias / gate projections.
// Block = 16 rows. LN writes the f16 tile straight into fragment-swizzled
// LDS; 8 waves each own 4 of 32 N-tiles. Outputs:
//   q_ws : f16, A-fragment order per (i,h,jt) tile     [ih][jt][512]
//   k_ws : f16, [ih][j][c]   (B fragment = contiguous in c)
//   v_ws : f16, [ih][c][j]   (B fragment = contiguous in j)
//   gate : f16, [row][128]
//   bias : f32, [h][j][k]  (k minor -> stage-2 reads are lane-contiguous)
// ---------------------------------------------------------------------------
__global__ __launch_bounds__(256) void ln_proj_kernel(
    const float* __restrict__ x,     const float* __restrict__ ln_g,
    const float* __restrict__ ln_b,  const _Float16* __restrict__ Wqt,
    const float* __restrict__ W_bias,const _Float16* __restrict__ Wgt,
    const float* __restrict__ b_gate,
    _Float16* __restrict__ q_ws, _Float16* __restrict__ k_ws,
    _Float16* __restrict__ v_ws, _Float16* __restrict__ gate_ws,
    float* __restrict__ bias_ws)
{
    __shared__ alignas(32) _Float16 lxF[4 * 32 * 16];   // frag-swizzled 16x128

    const int row0 = blockIdx.x * 16;            // global row (= i*256 + j)
    const int wave = threadIdx.x >> 5;
    const int lane = threadIdx.x & 31;

    // ---- LayerNorm: each wave normalizes 2 rows (32 lanes x 4 elems) ----
    #pragma unroll
    for (int rr = 0; rr < 2; ++rr) {
        const int r = wave * 2 + rr;
        const float* xr = x + (size_t)(row0 + r) * CIN;
        float v0 = xr[lane], v1 = xr[lane + 32], v2 = xr[lane + 64], v3 = xr[lane + 96];
        float s = v0 + v1 + v2 + v3;
        #pragma unroll
        for (int off = 16; off; off >>= 1) s += __shfl_xor(s, off, 32);
        const float mu = s * (1.0f / 128.0f);
        const float d0 = v0 - mu, d1 = v1 - mu, d2 = v2 - mu, d3 = v3 - mu;
        float s2 = d0 * d0 + d1 * d1 + d2 * d2 + d3 * d3;
        #pragma unroll
        for (int off = 16; off; off >>= 1) s2 += __shfl_xor(s2, off, 32);
        const float rs = rsqrtf(s2 * (1.0f / 128.0f) + LN_EPS);
        lxF[frag_idx(r, lane)]      = (_Float16)(d0 * rs * ln_g[lane]      + ln_b[lane]);
        lxF[frag_idx(r, lane + 32)] = (_Float16)(d1 * rs * ln_g[lane + 32] + ln_b[lane + 32]);
        lxF[frag_idx(r, lane + 64)] = (_Float16)(d2 * rs * ln_g[lane + 64] + ln_b[lane + 64]);
        lxF[frag_idx(r, lane + 96)] = (_Float16)(d3 * rs * ln_g[lane + 96] + ln_b[lane + 96]);
    }
    __syncthreads();

    // ---- A fragments: one aligned 32B LDS read per K chunk ----
    v16h afr[4];
    #pragma unroll
    for (int kk = 0; kk < 4; ++kk)
        afr[kk] = *(const v16h*)(&lxF[(kk * 32 + lane) * 16]);

    const int nlane = lane & 15;
    const int m0    = (lane < 16) ? 0 : 8;
    const int i     = row0 >> 8;
    const int jt    = (row0 >> 4) & 15;
    const int jb    = row0 & 255;                // j of first row in tile

    #pragma unroll
    for (int ntl = 0; ntl < 4; ++ntl) {
        const int col0    = (wave + ntl * 8) * 16;       // 0..496
        const bool isgate = (col0 >= 384);
        const _Float16* W = isgate ? Wgt : Wqt;
        const int ncol    = (isgate ? col0 - 384 : col0) + nlane;

        v8f c = {};
        #pragma unroll
        for (int kk = 0; kk < 4; ++kk)
            c = wmma16(afr[kk], load_b_frag_kcontig(W, CIN, kk * 32, ncol, lane), c);

        // ---- store: the (q|k|v|gate, c, h) routing is uniform per lane ----
        if (isgate) {
            const float bg = b_gate[ncol];
            #pragma unroll
            for (int r = 0; r < 8; ++r) {
                const float z = c[r] + bg;
                gate_ws[(size_t)(row0 + m0 + r) * (CATT * NH) + ncol] =
                    (_Float16)(1.0f / (1.0f + __expf(-z)));
            }
        } else {
            const int jcol = col0 + nlane;               // qkv column
            const int cch = jcol / 12, s = jcol % 12;
            const int h = s & 3, sel = s >> 2;
            const size_t ih = (size_t)(i * NH + h);
            if (sel == 0) {          // q: fragment order per (ih, jt) tile
                _Float16* qp = q_ws + (ih * 16 + jt) * 512;
                #pragma unroll
                for (int r = 0; r < 8; ++r)
                    qp[frag_idx(m0 + r, cch)] = (_Float16)c[r];
            } else if (sel == 1) {   // k: [ih][j][c]
                _Float16* kp = k_ws + ih * NSEQ * CATT;
                #pragma unroll
                for (int r = 0; r < 8; ++r)
                    kp[(size_t)(jb + m0 + r) * CATT + cch] = (_Float16)c[r];
            } else {                 // v: [ih][c][j]
                _Float16* vp = v_ws + (ih * CATT + cch) * NSEQ;
                #pragma unroll
                for (int r = 0; r < 8; ++r)
                    vp[jb + m0 + r] = (_Float16)c[r];
            }
        }
    }

    // ---- bias = x @ W_bias (16 rows x 4 heads), scalar dots.
    //      Stored transposed: bias_ws[h][y = j'][x = i']  (x minor). ----
    if (threadIdx.x < 64) {
        const int r = threadIdx.x >> 2, h = threadIdx.x & 3;
        float acc = 0.f;
        #pragma unroll 8
        for (int cc = 0; cc < CIN; ++cc)
            acc += (float)lxF[frag_idx(r, cc)] * W_bias[cc * NH + h];
        bias_ws[((size_t)h * NSEQ + (jb + r)) * NSEQ + i] = acc;
    }
}

// ---------------------------------------------------------------------------
// Stage 2: attention for one (i, h, j-tile). Block = 64 threads (2 waves).
// scores(16x256) = q_tile . k^T (one WMMA per 16-wide k tile, fragments are
// single 32B loads), + bias[h][j][k] (lane-contiguous), softmax over k (f32
// in LDS), normalized probabilities written to LDS in fragment order, then
// O = P @ V with one 32B A and one 32B B load per WMMA. Gated f16 store.
// ---------------------------------------------------------------------------
__global__ __launch_bounds__(64) void attn_kernel(
    const _Float16* __restrict__ q_ws, const _Float16* __restrict__ k_ws,
    const _Float16* __restrict__ v_ws, const _Float16* __restrict__ gate_ws,
    const float* __restrict__ bias_ws, _Float16* __restrict__ att_ws)
{
    __shared__ float S[16][NSEQ + 8];                  // raw scores (f32)
    __shared__ alignas(32) _Float16 Pf[8 * 32 * 16];   // probs, frag order

    const int bid  = blockIdx.x;
    const int jt   = bid & 15;
    const int h    = (bid >> 4) & 3;
    const int i    = bid >> 6;
    const int wave = threadIdx.x >> 5;
    const int lane = threadIdx.x & 31;
    const float scale = 0.17677669529663687f;          // 1/sqrt(32)

    const size_t ih = (size_t)(i * NH + h);
    const int nlane = lane & 15;
    const int m0    = (lane < 16) ? 0 : 8;

    // q A-fragment: single 32B load (stored in fragment order by stage 1)
    const v16h aq = *(const v16h*)(q_ws + (ih * 16 + jt) * 512 + lane * 16);

    // bias rows for this block: bias_ws[h][jrow][k], k minor (contiguous).
    const float* biasrow = bias_ws + ((size_t)h * NSEQ + jt * 16) * NSEQ;

    // ---- scores: each wave does 8 of the 16 k tiles ----
    for (int kt = wave; kt < 16; kt += 2) {
        const int n0 = kt * 16;
        const int kofs = (lane < 16) ? 0 : 16;
        const v16h bk = *(const v16h*)(k_ws + (ih * NSEQ + n0 + nlane) * CATT + kofs);
        v8f c = {};
        c = wmma16(aq, bk, c);
        const int nn = n0 + nlane;
        #pragma unroll
        for (int r = 0; r < 8; ++r)
            S[m0 + r][nn] = c[r] * scale + biasrow[(size_t)(m0 + r) * NSEQ + nn];
    }
    __syncthreads();

    // ---- softmax over k (256 cols); wave w: rows 8w..8w+7.  Normalized
    //      probabilities land directly in fragment-order f16 LDS. ----
    for (int r = wave * 8; r < wave * 8 + 8; ++r) {
        float vals[8];
        float mx = -1e30f;
        #pragma unroll
        for (int u = 0; u < 8; ++u) {
            vals[u] = S[r][lane + u * 32];
            mx = fmaxf(mx, vals[u]);
        }
        #pragma unroll
        for (int off = 16; off; off >>= 1) mx = fmaxf(mx, __shfl_xor(mx, off, 32));
        float sum = 0.f;
        #pragma unroll
        for (int u = 0; u < 8; ++u) { vals[u] = __expf(vals[u] - mx); sum += vals[u]; }
        #pragma unroll
        for (int off = 16; off; off >>= 1) sum += __shfl_xor(sum, off, 32);
        const float inv = 1.0f / sum;
        #pragma unroll
        for (int u = 0; u < 8; ++u)
            Pf[frag_idx(r, lane + u * 32)] = (_Float16)(vals[u] * inv);
    }
    __syncthreads();

    // ---- O = P @ V : wave w owns c tile w (cols 16w..16w+15), K = 256 ----
    const int ct = wave;
    v8f o = {};
    #pragma unroll
    for (int kk = 0; kk < 8; ++kk) {
        const v16h ap = *(const v16h*)(&Pf[(kk * 32 + lane) * 16]);
        const int kofs = kk * 32 + ((lane < 16) ? 0 : 16);
        const v16h bv = *(const v16h*)(v_ws + (ih * CATT + ct * 16 + nlane) * NSEQ + kofs);
        o = wmma16(ap, bv, o);
    }

    // ---- gate + store att[i][j][c*NH + h] (f16) ----
    {
        const int c = ct * 16 + nlane;
        #pragma unroll
        for (int r = 0; r < 8; ++r) {
            const int j = jt * 16 + m0 + r;
            const size_t idx =
                ((size_t)i * NSEQ + j) * (CATT * NH) + (size_t)c * NH + h;
            att_ws[idx] = (_Float16)(o[r] * (float)gate_ws[idx]);
        }
    }
}

// ---------------------------------------------------------------------------
// Stage 3: out = att @ W_out + b_out  (65536x128 @ 128x128), f32 out.
// ---------------------------------------------------------------------------
__global__ __launch_bounds__(256) void out_proj_kernel(
    const _Float16* __restrict__ att_ws, const _Float16* __restrict__ Wot,
    const float* __restrict__ b_out, float* __restrict__ out)
{
    __shared__ alignas(32) _Float16 laF[4 * 32 * 16];  // frag-swizzled 16x128

    const int row0 = blockIdx.x * 16;
    const int wave = threadIdx.x >> 5;
    const int lane = threadIdx.x & 31;

    for (int t = threadIdx.x; t < 16 * CIN; t += 256)
        laF[frag_idx(t >> 7, t & 127)] = att_ws[(size_t)row0 * CIN + t];
    __syncthreads();

    v16h afr[4];
    #pragma unroll
    for (int kk = 0; kk < 4; ++kk)
        afr[kk] = *(const v16h*)(&laF[(kk * 32 + lane) * 16]);

    const int n = wave * 16 + (lane & 15);
    v8f c = {};
    #pragma unroll
    for (int kk = 0; kk < 4; ++kk)
        c = wmma16(afr[kk], load_b_frag_kcontig(Wot, CIN, kk * 32, n, lane), c);

    const int m0 = (lane < 16) ? 0 : 8;
    const float bo = b_out[n];
    #pragma unroll
    for (int r = 0; r < 8; ++r)
        out[(size_t)(row0 + m0 + r) * CIN + n] = c[r] + bo;
}

// ---------------------------------------------------------------------------
extern "C" void kernel_launch(void* const* d_in, const int* in_sizes, int n_in,
                              void* d_out, int out_size, void* d_ws, size_t ws_size,
                              hipStream_t stream) {
    const float* x      = (const float*)d_in[0];
    const float* ln_g   = (const float*)d_in[1];
    const float* ln_b   = (const float*)d_in[2];
    const float* W_qkv  = (const float*)d_in[3];
    const float* W_bias = (const float*)d_in[4];
    const float* W_gate = (const float*)d_in[5];
    const float* b_gate = (const float*)d_in[6];
    const float* W_out  = (const float*)d_in[7];
    const float* b_out  = (const float*)d_in[8];
    float* out = (float*)d_out;

    // Workspace carve-up. PER = N*N*CATT*NH elements = 8,388,608.
    const size_t PER = (size_t)NSEQ * NSEQ * CATT * NH;
    char* p = (char*)d_ws;
    _Float16* q_ws    = (_Float16*)p; p += PER * sizeof(_Float16);
    _Float16* k_ws    = (_Float16*)p; p += PER * sizeof(_Float16);
    _Float16* v_ws    = (_Float16*)p; p += PER * sizeof(_Float16);
    _Float16* gate_ws = (_Float16*)p; p += PER * sizeof(_Float16);
    _Float16* att_ws  = (_Float16*)p; p += PER * sizeof(_Float16);
    float*    bias_ws = (float*)p;    p += (size_t)NH * NSEQ * NSEQ * sizeof(float);
    _Float16* Wqt     = (_Float16*)p; p += (size_t)384 * CIN * sizeof(_Float16);
    _Float16* Wgt     = (_Float16*)p; p += (size_t)128 * CIN * sizeof(_Float16);
    _Float16* Wot     = (_Float16*)p; p += (size_t)128 * CIN * sizeof(_Float16);

    const int mtiles = (NSEQ * NSEQ) / 16;                // 4096

    prep_weights_kernel<<<(640 * CIN + 255) / 256, 256, 0, stream>>>(
        W_qkv, W_gate, W_out, Wqt, Wgt, Wot);

    ln_proj_kernel<<<mtiles, 256, 0, stream>>>(
        x, ln_g, ln_b, Wqt, W_bias, Wgt, b_gate,
        q_ws, k_ws, v_ws, gate_ws, bias_ws);

    attn_kernel<<<NSEQ * NH * (NSEQ / 16), 64, 0, stream>>>(
        q_ws, k_ws, v_ws, gate_ws, bias_ws, att_ws);

    out_proj_kernel<<<mtiles, 256, 0, stream>>>(att_ws, Wot, b_out, out);
}
